// AutoEncoder_51642686767592
// MI455X (gfx1250) — matrix operand order repaired and checked by reading, not compiled, not run on hardware
//
#include <hip/hip_runtime.h>
#include <hip/hip_bf16.h>
#include <math.h>

// ---------------------------------------------------------------------------
// CDNA5 / gfx1250 WMMA types
// ---------------------------------------------------------------------------
typedef __attribute__((ext_vector_type(16))) __bf16 v16bf;
typedef __attribute__((ext_vector_type(8)))  float  v8f;

#define B_ROWS 16384
#define D_COLS 4096
#define BD_ELEMS ((size_t)B_ROWS * (size_t)D_COLS)

// ---------------------------------------------------------------------------
// Workspace layout (bytes)
// ---------------------------------------------------------------------------
#define OFF_S      0                               // s[16384] f32
#define OFF_RS     (64*1024)                       // rowscale[16384] f32
#define OFF_SCAL   (128*1024)                      // scalars: f32[16], then double[2] at +128
#define OFF_CS1    (OFF_SCAL + 256)                // colsum1[64] + colsq1[64]
#define OFF_CS2    (OFF_CS1 + 512)                 // colsum2[32] + colsq2[32]
#define OFF_CS3    (OFF_CS2 + 256)                 // colsum3[64] + colsq3[64]
#define OFF_SS1    (OFF_CS3 + 512)                 // S1[64], T1[64]
#define OFF_SS2    (OFF_SS1 + 512)                 // S2[32], T2[32]
#define OFF_SS3    (OFF_SS2 + 256)                 // S3[64], T3[64]
#define OFF_H1     (144*1024)                      // h1raw 16384x64 f32 (4MB)
#define OFF_H2     (OFF_H1 + B_ROWS*64*4)          // h2raw 16384x32 f32 (2MB)
#define OFF_H3     (OFF_H2 + B_ROWS*32*4)          // h3raw 16384x64 f32 (4MB)
#define OFF_PWIN   (OFF_H3 + B_ROWS*64*4)          // packed bf16 W_in  (4096x64)
#define OFF_PWENC  (OFF_PWIN + 4096*64*2)          // packed bf16 W_enc (64x32)
#define OFF_PWDEC  (OFF_PWENC + 64*32*2)           // packed bf16 W_dec (32x64)
#define OFF_PWPI   (OFF_PWDEC + 32*64*2)           // packed bf16 W_pi  (64x4096)
#define OFF_PWM    (OFF_PWPI + 64*4096*2)          // (pwpi/pwm/pwth contiguous!)
#define OFF_PWTH   (OFF_PWM  + 64*4096*2)

// ---------------------------------------------------------------------------
// Helpers
// ---------------------------------------------------------------------------
static __device__ inline unsigned short f2bf_bits(float f) {
    union { __bf16 b; unsigned short u; } c;
    c.b = (__bf16)f;                      // hardware f32->bf16 convert (RNE)
    return c.u;
}

static __device__ inline v16bf load_pw(const unsigned short* p) {
    union { uint4 q[2]; v16bf v; } u;
    const uint4* q = (const uint4*)p;     // 32B-aligned fragment
    u.q[0] = q[0];
    u.q[1] = q[1];
    return u.v;
}

static __device__ inline v16bf pack_bf16(const float* v) {
    v16bf a;
#pragma unroll
    for (int j = 0; j < 16; ++j) a[j] = (__bf16)v[j];   // native cvt, not soft-ALU
    return a;
}

// Load the 16 floats this lane owns of a 16x32 A-tile (K runs 8h..8h+7, 16+8h..+7)
static __device__ inline void load16(const float* p, int k0, float* out) {
    float4 a = *(const float4*)(p + k0);
    float4 b = *(const float4*)(p + k0 + 4);
    float4 c = *(const float4*)(p + k0 + 16);
    float4 d = *(const float4*)(p + k0 + 20);
    out[0]=a.x; out[1]=a.y; out[2]=a.z;  out[3]=a.w;
    out[4]=b.x; out[5]=b.y; out[6]=b.z;  out[7]=b.w;
    out[8]=c.x; out[9]=c.y; out[10]=c.z; out[11]=c.w;
    out[12]=d.x; out[13]=d.y; out[14]=d.z; out[15]=d.w;
}

static __device__ inline v8f wmma_bf16(v16bf a, v16bf b, v8f c) {
    // D = A(16x32 bf16) x B(32x16 bf16) + C(16x16 f32)
    return __builtin_amdgcn_wmma_f32_16x16x32_bf16(false, a, false, b, (short)0, c,
                                                   false, false);
}

// A-fragment = relu(raw*S + T) (fused BatchNorm affine) converted to bf16
static __device__ inline v16bf act_frag(const float* ir, const float* S, const float* T,
                                        int kbase, int lh) {
    int k0 = kbase + 8 * lh;
    float r[16], sc[16], sh[16], v[16];
    load16(ir, k0, r);
    load16(S,  k0, sc);
    load16(T,  k0, sh);
#pragma unroll
    for (int j = 0; j < 16; ++j) v[j] = fmaxf(fmaf(r[j], sc[j], sh[j]), 0.0f);
    return pack_bf16(v);
}

// ---------------------------------------------------------------------------
// Preprocessing kernels
// ---------------------------------------------------------------------------
__global__ __launch_bounds__(256) void k_init(float* p, int n) {
    int i = blockIdx.x * 256 + threadIdx.x;
    if (i < n) p[i] = 0.0f;
}

// Pack an (K x N) f32 weight into the WMMA-B bf16 fragment layout.
__global__ __launch_bounds__(256) void k_pack(const float* W, unsigned short* out,
                                              int K, int N) {
    int e = blockIdx.x * 256 + threadIdx.x;
    if (e >= K * N) return;
    int NT = N >> 4;
    int j    = e & 15;
    int lane = (e >> 4) & 31;
    int rest = e >> 9;
    int t    = rest % NT;
    int kb   = rest / NT;
    int k = kb * 32 + 8 * (lane >> 4) + j + ((j >> 3) << 3);   // lanes 0-15: K 0..7,16..23, lanes 16-31: K 8..15,24..31
    int n = t * 16 + (lane & 15);
    out[e] = f2bf_bits(W[(size_t)k * N + n]);
}

__global__ __launch_bounds__(256) void k_rowsum(const float* x, float* s) {
    __shared__ float red[256];
    const float* xr = x + (size_t)blockIdx.x * D_COLS;
    float a = 0.0f;
    for (int j = threadIdx.x; j < D_COLS; j += 256) a += xr[j];
    red[threadIdx.x] = a;
    __syncthreads();
    for (int st = 128; st > 0; st >>= 1) {
        if (threadIdx.x < st) red[threadIdx.x] += red[threadIdx.x + st];
        __syncthreads();
    }
    if (threadIdx.x == 0) s[blockIdx.x] = red[0];
}

static __device__ inline unsigned int mapf(float f) {
    unsigned int u = __float_as_uint(f);
    return (u & 0x80000000u) ? ~u : (u | 0x80000000u);
}
static __device__ inline float unmapf(unsigned int k) {
    return (k & 0x80000000u) ? __uint_as_float(k ^ 0x80000000u) : __uint_as_float(~k);
}

// Exact lower-median via 4x8-bit radix selection; single block, deterministic.
__global__ __launch_bounds__(256) void k_median(const float* s, int n, float* scal) {
    __shared__ unsigned int hist[256];
    __shared__ unsigned int sh_pref, sh_mask, sh_k;
    int tid = threadIdx.x;
    if (tid == 0) { sh_pref = 0; sh_mask = 0; sh_k = (unsigned)((n - 1) / 2); }
    __syncthreads();
    for (int pass = 0; pass < 4; ++pass) {
        int shift = 24 - 8 * pass;
        hist[tid] = 0;
        __syncthreads();
        unsigned pref = sh_pref, mask = sh_mask;
        for (int i = tid; i < n; i += 256) {
            unsigned key = mapf(s[i]);
            if ((key & mask) == pref) atomicAdd(&hist[(key >> shift) & 255], 1u);
        }
        __syncthreads();
        if (tid == 0) {
            unsigned k = sh_k, cum = 0;
            int b = 0;
            for (; b < 256; ++b) {
                if (cum + hist[b] > k) break;
                cum += hist[b];
            }
            sh_k = k - cum;
            sh_pref = pref | ((unsigned)b << shift);
            sh_mask = mask | (0xFFu << shift);
        }
        __syncthreads();
    }
    if (tid == 0) scal[0] = unmapf(sh_pref);
}

__global__ __launch_bounds__(256) void k_rowscale(const float* s, const float* scal,
                                                  float* rowscale) {
    int i = blockIdx.x * 256 + threadIdx.x;
    if (i < B_ROWS) rowscale[i] = scal[0] / s[i];   // med / s_i
}

// Global sum / sumsq of norm = log(x * rowscale + 1); f64 atomics for the tail.
__global__ __launch_bounds__(256) void k_stats(const float* x, const float* rowscale,
                                               double* dscal) {
    __shared__ float rs[256], rq[256];
    size_t base = (size_t)blockIdx.x * 32768;
    float a = 0.0f, b = 0.0f;
    for (int i = threadIdx.x; i < 32768; i += 256) {
        size_t e = base + i;
        float v = logf(fmaf(x[e], rowscale[e >> 12], 1.0f));
        a += v;
        b += v * v;
    }
    rs[threadIdx.x] = a; rq[threadIdx.x] = b;
    __syncthreads();
    for (int st = 128; st > 0; st >>= 1) {
        if (threadIdx.x < st) {
            rs[threadIdx.x] += rs[threadIdx.x + st];
            rq[threadIdx.x] += rq[threadIdx.x + st];
        }
        __syncthreads();
    }
    if (threadIdx.x == 0) {
        atomicAdd(&dscal[0], (double)rs[0]);
        atomicAdd(&dscal[1], (double)rq[0]);
    }
}

__global__ void k_statsfin(const double* dscal, float* scal) {
    double N = (double)BD_ELEMS;
    double sum = dscal[0], sq = dscal[1];
    double mean = sum / N;
    double var = (sq - sum * sum / N) / (N - 1.0);   // ddof=1, as torch .std()
    scal[3] = (float)mean;
    scal[4] = (float)(1.0 / sqrt(var));
}

// Per-column BN fold: act = relu(raw*S + T). Linear bias cancels inside BN.
__global__ void k_bnfin(const float* cs, const float* g, const float* bt,
                        float* ST, int H) {
    int k = threadIdx.x;
    if (k >= H) return;
    float invB = 1.0f / (float)B_ROWS;
    float mean = cs[k] * invB;
    float var  = cs[H + k] * invB - mean * mean;     // biased batch variance
    float s = g[k] * rsqrtf(var + 1e-5f);
    ST[k] = s;
    ST[H + k] = fmaf(-mean, s, bt[k]);
}

// ---------------------------------------------------------------------------
// GEMM 1: h1raw = standardized(log(x*rsc+1)) @ W_in    (16384x4096 x 4096x64)
// One wave handles a 32x64 output tile (two 16-row A fragments share every B
// fragment -> 2 WMMAs per B load, half the weight traffic per row).
// ---------------------------------------------------------------------------
__global__ __launch_bounds__(128) void k_gemm1(const float* x, const unsigned short* pw,
                                               const float* rowscale, const float* scal,
                                               float* h1raw, float* cs1) {
    __shared__ float lsum[64], lsq[64];
    int tid = threadIdx.x;
    if (tid < 64) { lsum[tid] = 0.0f; lsq[tid] = 0.0f; }
    __syncthreads();
    int wid = tid >> 5, lane = tid & 31;
    int lh = lane >> 4, lm = lane & 15;
    int wt = blockIdx.x * 4 + wid;            // 0..511, 32 rows each
    int row0 = wt * 32;
    const float* xr0 = x + (size_t)(row0 + lm) * D_COLS;
    const float* xr1 = x + (size_t)(row0 + 16 + lm) * D_COLS;
    float rsc0 = rowscale[row0 + lm];
    float rsc1 = rowscale[row0 + 16 + lm];
    float mean = scal[3], rstd = scal[4];

    v8f zero = {0.f,0.f,0.f,0.f,0.f,0.f,0.f,0.f};
    v8f accA[4], accB[4];
#pragma unroll
    for (int t = 0; t < 4; ++t) { accA[t] = zero; accB[t] = zero; }

    for (int kb = 0; kb < 128; ++kb) {
        int kk = kb * 32;
        if (kb + 8 < 128) {
            __builtin_prefetch(xr0 + kk + 256, 0, 1);
            __builtin_prefetch(xr1 + kk + 256, 0, 1);
        }
        float r0[16], r1[16], v0[16], v1[16];
        load16(xr0, kk + 8 * lh, r0);
        load16(xr1, kk + 8 * lh, r1);
#pragma unroll
        for (int j = 0; j < 16; ++j) {
            v0[j] = (logf(fmaf(r0[j], rsc0, 1.0f)) - mean) * rstd;
            v1[j] = (logf(fmaf(r1[j], rsc1, 1.0f)) - mean) * rstd;
        }
        v16bf a0 = pack_bf16(v0);
        v16bf a1 = pack_bf16(v1);
#pragma unroll
        for (int t = 0; t < 4; ++t) {
            v16bf b = load_pw(pw + ((size_t)(kb * 4 + t) * 32 + lane) * 16);
            accA[t] = wmma_bf16(a0, b, accA[t]);
            accB[t] = wmma_bf16(a1, b, accB[t]);
        }
    }

#pragma unroll
    for (int t = 0; t < 4; ++t) {
        float ps = 0.0f, pq = 0.0f;
#pragma unroll
        for (int v = 0; v < 8; ++v) {
            int m = v + 8 * lh;
            float va = accA[t][v];
            float vb = accB[t][v];
            h1raw[(size_t)(row0 + m) * 64 + t * 16 + lm] = va;
            h1raw[(size_t)(row0 + 16 + m) * 64 + t * 16 + lm] = vb;
            ps += va + vb;
            pq += va * va + vb * vb;
        }
        atomicAdd(&lsum[t * 16 + lm], ps);
        atomicAdd(&lsq[t * 16 + lm], pq);
    }
    __syncthreads();
    if (tid < 64) {
        atomicAdd(&cs1[tid], lsum[tid]);
        atomicAdd(&cs1[64 + tid], lsq[tid]);
    }
}

// ---------------------------------------------------------------------------
// Mid GEMMs: outraw = relu(inraw*S+T) @ W     (templated on K/32 and N/16)
// ---------------------------------------------------------------------------
template <int KB, int NT>
__global__ __launch_bounds__(128) void k_mid(const float* inraw, const float* S,
                                             const float* T, const unsigned short* pw,
                                             float* outraw, float* cs) {
    const int Kdim = KB * 32, Ndim = NT * 16;
    __shared__ float lsum[64], lsq[64];
    int tid = threadIdx.x;
    if (tid < 64) { lsum[tid] = 0.0f; lsq[tid] = 0.0f; }
    __syncthreads();
    int wid = tid >> 5, lane = tid & 31;
    int lh = lane >> 4, lm = lane & 15;
    int rowtile = blockIdx.x * 4 + wid;
    const float* ir = inraw + (size_t)(rowtile * 16 + lm) * Kdim;

    v8f zero = {0.f,0.f,0.f,0.f,0.f,0.f,0.f,0.f};
    v8f acc[NT];
#pragma unroll
    for (int t = 0; t < NT; ++t) acc[t] = zero;

#pragma unroll
    for (int kb = 0; kb < KB; ++kb) {
        v16bf a = act_frag(ir, S, T, kb * 32, lh);
#pragma unroll
        for (int t = 0; t < NT; ++t) {
            v16bf b = load_pw(pw + ((size_t)(kb * NT + t) * 32 + lane) * 16);
            acc[t] = wmma_bf16(a, b, acc[t]);
        }
    }

#pragma unroll
    for (int t = 0; t < NT; ++t) {
        float ps = 0.0f, pq = 0.0f;
#pragma unroll
        for (int v = 0; v < 8; ++v) {
            float val = acc[t][v];
            int m = v + 8 * lh;
            outraw[(size_t)(rowtile * 16 + m) * Ndim + t * 16 + lm] = val;
            ps += val;
            pq += val * val;
        }
        atomicAdd(&lsum[t * 16 + lm], ps);
        atomicAdd(&lsq[t * 16 + lm], pq);
    }
    __syncthreads();
    if (tid < Ndim) {
        atomicAdd(&cs[tid], lsum[tid]);
        atomicAdd(&cs[Ndim + tid], lsq[tid]);
    }
}

// ---------------------------------------------------------------------------
// Output GEMMs: PI = sigmoid(h3 @ W_pi + b), M/THETA = exp(h3 @ W + b)
// Each block stages its column-block's 24KB of weight fragments (3 matrices x
// 2 K-steps x 4 N-tiles) into LDS ONCE via the CDNA5 async global->LDS path,
// then 4 waves x 4 row-tiles consume them with ds_load_b128.
// ---------------------------------------------------------------------------
static __device__ inline v16bf load_lds_frag(const unsigned char* smem, int c, int t,
                                             int lane) {
    const uint4* p = (const uint4*)(smem + c * 4096 + (t * 32 + lane) * 32);
    union { uint4 q[2]; v16bf v; } u;
    u.q[0] = p[0];
    u.q[1] = p[1];
    return u.v;
}

static __device__ inline void out_block(const unsigned char* smem, int m,
                                        const float* bias, float* outp,
                                        v16bf a0, v16bf a1, int cb,
                                        int lane, int lh, int lm, int row0, int mode) {
    v8f zero = {0.f,0.f,0.f,0.f,0.f,0.f,0.f,0.f};
#pragma unroll
    for (int t = 0; t < 4; ++t) {
        v8f acc = zero;
        acc = wmma_bf16(a0, load_lds_frag(smem, m * 2 + 0, t, lane), acc);
        acc = wmma_bf16(a1, load_lds_frag(smem, m * 2 + 1, t, lane), acc);
        int col = cb * 64 + t * 16 + lm;
        float bs = bias[col];
#pragma unroll
        for (int v = 0; v < 8; ++v) {
            float z = acc[v] + bs;
            z = mode ? expf(z) : 1.0f / (1.0f + expf(-z));
            outp[(size_t)(row0 + v + 8 * lh) * D_COLS + col] = z;
        }
    }
}

__global__ __launch_bounds__(128) void k_out(const float* h3raw, const float* S3,
                                             const float* T3,
                                             const unsigned short* pw3,  // pwpi base
                                             const float* bpi, const float* bm,
                                             const float* bth, float* out) {
    __shared__ unsigned char smem[24576];   // 6 chunks x 4KB
    int tid = threadIdx.x;
    int wid = tid >> 5, lane = tid & 31;
    int lh = lane >> 4, lm = lane & 15;
    int cb = blockIdx.x;

    // ---- stage 24KB of weight fragments for this column block into LDS ----
    // chunk c = m*2+ks is 4KB contiguous in both global and LDS.
#pragma unroll
    for (int r = 0; r < 12; ++r) {
        int u = tid + 128 * r;                 // 16B unit 0..1535
        int c = u >> 8;                        // 0..5
        int w = u & 255;
        int m = c >> 1, ks = c & 1;
        const unsigned short* gsrc =
            pw3 + (size_t)m * 262144 + (size_t)ks * 131072 + (size_t)cb * 2048
                + (size_t)w * 8;
        unsigned int ldsOff = (unsigned int)(size_t)(smem + u * 16);
        unsigned long long ga = (unsigned long long)(size_t)gsrc;
        asm volatile("global_load_async_to_lds_b128 %0, %1, off"
                     :: "v"(ldsOff), "v"(ga) : "memory");
    }
    asm volatile("s_wait_asynccnt 0x0" ::: "memory");
    __syncthreads();

    // ---- 4 row-tiles per wave ----
#pragma unroll
    for (int it = 0; it < 4; ++it) {
        int rowtile = blockIdx.y * 16 + wid * 4 + it;
        int row0 = rowtile * 16;
        const float* ir = h3raw + (size_t)(row0 + lm) * 64;
        v16bf a0 = act_frag(ir, S3, T3, 0, lh);
        v16bf a1 = act_frag(ir, S3, T3, 32, lh);
        out_block(smem, 0, bpi, out,                a0, a1, cb, lane, lh, lm, row0, 0);
        out_block(smem, 1, bm,  out + BD_ELEMS,     a0, a1, cb, lane, lh, lm, row0, 1);
        out_block(smem, 2, bth, out + 2 * BD_ELEMS, a0, a1, cb, lane, lh, lm, row0, 1);
    }
}

// ---------------------------------------------------------------------------
// Host launcher
// ---------------------------------------------------------------------------
extern "C" void kernel_launch(void* const* d_in, const int* in_sizes, int n_in,
                              void* d_out, int out_size, void* d_ws, size_t ws_size,
                              hipStream_t stream) {
    (void)in_sizes; (void)n_in; (void)out_size; (void)ws_size;
    const float* x     = (const float*)d_in[0];
    const float* W_in  = (const float*)d_in[1];
    const float* g1    = (const float*)d_in[3];
    const float* bt1   = (const float*)d_in[4];
    const float* W_enc = (const float*)d_in[5];
    const float* g2    = (const float*)d_in[7];
    const float* bt2   = (const float*)d_in[8];
    const float* W_dec = (const float*)d_in[9];
    const float* g3    = (const float*)d_in[11];
    const float* bt3   = (const float*)d_in[12];
    const float* W_pi  = (const float*)d_in[13];
    const float* b_pi  = (const float*)d_in[14];
    const float* W_m   = (const float*)d_in[15];
    const float* b_m   = (const float*)d_in[16];
    const float* W_th  = (const float*)d_in[17];
    const float* b_th  = (const float*)d_in[18];
    float* out = (float*)d_out;

    char* ws = (char*)d_ws;
    float*  s        = (float*)(ws + OFF_S);
    float*  rowscale = (float*)(ws + OFF_RS);
    float*  scal     = (float*)(ws + OFF_SCAL);
    double* dscal    = (double*)(ws + OFF_SCAL + 128);
    float*  cs1      = (float*)(ws + OFF_CS1);
    float*  cs2      = (float*)(ws + OFF_CS2);
    float*  cs3      = (float*)(ws + OFF_CS3);
    float*  ss1      = (float*)(ws + OFF_SS1);
    float*  ss2      = (float*)(ws + OFF_SS2);
    float*  ss3      = (float*)(ws + OFF_SS3);
    float*  h1raw    = (float*)(ws + OFF_H1);
    float*  h2raw    = (float*)(ws + OFF_H2);
    float*  h3raw    = (float*)(ws + OFF_H3);
    unsigned short* pwin  = (unsigned short*)(ws + OFF_PWIN);
    unsigned short* pwenc = (unsigned short*)(ws + OFF_PWENC);
    unsigned short* pwdec = (unsigned short*)(ws + OFF_PWDEC);
    unsigned short* pwpi  = (unsigned short*)(ws + OFF_PWPI);
    unsigned short* pwm   = (unsigned short*)(ws + OFF_PWM);
    unsigned short* pwth  = (unsigned short*)(ws + OFF_PWTH);

    // 0) zero scalar + column-stat accumulators (scal..cs3 = 384 f32, incl. doubles)
    hipLaunchKernelGGL(k_init, dim3(2), dim3(256), 0, stream, scal, 384);

    // 1) pack weights to WMMA-B bf16 fragment layout
    hipLaunchKernelGGL(k_pack, dim3(1024), dim3(256), 0, stream, W_in,  pwin,  4096, 64);
    hipLaunchKernelGGL(k_pack, dim3(8),    dim3(256), 0, stream, W_enc, pwenc, 64,   32);
    hipLaunchKernelGGL(k_pack, dim3(8),    dim3(256), 0, stream, W_dec, pwdec, 32,   64);
    hipLaunchKernelGGL(k_pack, dim3(1024), dim3(256), 0, stream, W_pi,  pwpi,  64, 4096);
    hipLaunchKernelGGL(k_pack, dim3(1024), dim3(256), 0, stream, W_m,   pwm,   64, 4096);
    hipLaunchKernelGGL(k_pack, dim3(1024), dim3(256), 0, stream, W_th,  pwth,  64, 4096);

    // 2) preprocessing statistics
    hipLaunchKernelGGL(k_rowsum,   dim3(B_ROWS), dim3(256), 0, stream, x, s);
    hipLaunchKernelGGL(k_median,   dim3(1),      dim3(256), 0, stream, s, B_ROWS, scal);
    hipLaunchKernelGGL(k_rowscale, dim3(64),     dim3(256), 0, stream, s, scal, rowscale);
    hipLaunchKernelGGL(k_stats,    dim3(2048),   dim3(256), 0, stream, x, rowscale, dscal);
    hipLaunchKernelGGL(k_statsfin, dim3(1),      dim3(1),   0, stream, dscal, scal);

    // 3) encoder/decoder funnel (WMMA bf16, BN stats fused)
    hipLaunchKernelGGL(k_gemm1, dim3(128), dim3(128), 0, stream,
                       x, pwin, rowscale, scal, h1raw, cs1);
    hipLaunchKernelGGL(k_bnfin, dim3(1), dim3(64), 0, stream, cs1, g1, bt1, ss1, 64);
    hipLaunchKernelGGL((k_mid<2, 2>), dim3(256), dim3(128), 0, stream,
                       h1raw, ss1, ss1 + 64, pwenc, h2raw, cs2);
    hipLaunchKernelGGL(k_bnfin, dim3(1), dim3(64), 0, stream, cs2, g2, bt2, ss2, 32);
    hipLaunchKernelGGL((k_mid<1, 4>), dim3(256), dim3(128), 0, stream,
                       h2raw, ss2, ss2 + 32, pwdec, h3raw, cs3);
    hipLaunchKernelGGL(k_bnfin, dim3(1), dim3(64), 0, stream, cs3, g3, bt3, ss3, 64);

    // 4) three fat output GEMMs + sigmoid/exp epilogues (LDS-staged weights)
    hipLaunchKernelGGL(k_out, dim3(64, 64), dim3(128), 0, stream,
                       h3raw, ss3, ss3 + 64, pwpi, b_pi, b_m, b_th, out);
}
